// GCN_35545149342389
// MI455X (gfx1250) — matrix operand order repaired and checked by reading, not compiled, not run on hardware
//
#include <hip/hip_runtime.h>
#include <hip/hip_bf16.h>
#include <math.h>

#define N_NODES 11473
#define F_FEAT  128
#define HID     32
#define NCLS    10

typedef __attribute__((ext_vector_type(16))) _Float16 v16h;
typedef __attribute__((ext_vector_type(8)))  _Float16 v8h;
typedef __attribute__((ext_vector_type(8)))  float    v8f;

// Load a 16-half WMMA fragment from two contiguous 16-byte LDS chunks.
__device__ __forceinline__ v16h load_frag(const _Float16* p0, const _Float16* p1) {
    v8h lo = *(const v8h*)p0;
    v8h hi = *(const v8h*)p1;
    v16h r;
#pragma unroll
    for (int i = 0; i < 8; ++i) { r[i] = lo[i]; r[i + 8] = hi[i]; }
    return r;
}

__device__ __forceinline__ float sigm_f(float x) {
    return 1.0f / (1.0f + __expf(-x));
}
__device__ __forceinline__ float tanh_f(float x) {
    return 2.0f / (1.0f + __expf(-2.0f * x)) - 1.0f;
}

// ---------------------------------------------------------------------------
// Fused 2-layer LSTM (seq_len=128, hidden=32) + MLP + log_softmax.
// 1 block = 256 threads = 8 wave32s; each wave owns a 16-row tile of nodes
// and runs all 128 timesteps, with per-step WMMA f32_16x16x32_f16 GEMMs.
// ---------------------------------------------------------------------------
__global__ __launch_bounds__(256) void GCN_lstm_mlp_kernel(
    const float* __restrict__ x,
    const float* __restrict__ Wih0, const float* __restrict__ Whh0,
    const float* __restrict__ bih0, const float* __restrict__ bhh0,
    const float* __restrict__ Wih1, const float* __restrict__ Whh1,
    const float* __restrict__ bih1, const float* __restrict__ bhh1,
    const float* __restrict__ A1, const float* __restrict__ b1,
    const float* __restrict__ A2, const float* __restrict__ b2,
    const float* __restrict__ A3, const float* __restrict__ b3,
    float* __restrict__ logp)
{
    // LDS: gate weights as f16 row-major [g=0..127][k=0..31] -> B fragments are
    // two contiguous ds_load_b128 per lane. Plus per-wave h staging / hsum.
    __shared__ _Float16 sWhh0[128 * 32];
    __shared__ _Float16 sWih1[128 * 32];
    __shared__ _Float16 sWhh1[128 * 32];
    __shared__ float    sMlp[32 * 32 + 32 + 32 * 16 + 16 + 16 * 10 + 10]; // 1754 f32
    __shared__ float    sH[8][512];   // per-wave scratch: f16 h0/h1 staging, then f32 hsum

    const int tid = threadIdx.x;

    for (int i = tid; i < 128 * 32; i += 256) {
        sWhh0[i] = (_Float16)Whh0[i];
        sWih1[i] = (_Float16)Wih1[i];
        sWhh1[i] = (_Float16)Whh1[i];
    }
    for (int i = tid; i < 1024; i += 256) sMlp[i]        = A1[i];
    for (int i = tid; i < 32;   i += 256) sMlp[1024 + i] = b1[i];
    for (int i = tid; i < 512;  i += 256) sMlp[1056 + i] = A2[i];
    for (int i = tid; i < 16;   i += 256) sMlp[1568 + i] = b2[i];
    for (int i = tid; i < 160;  i += 256) sMlp[1584 + i] = A3[i];
    for (int i = tid; i < 10;   i += 256) sMlp[1744 + i] = b3[i];
    __syncthreads();

    const int wave = tid >> 5;
    const int lane = tid & 31;
    const int lo   = lane & 15;   // N index within 16x16 tile (C/D, B layouts)
    const int hi   = lane >> 4;   // half-select
    const int row0 = blockIdx.x * 128 + wave * 16;

    _Float16* h0stg = (_Float16*)&sH[wave][0];     // 512 halves (16x32 f16)
    _Float16* h1stg = (_Float16*)&sH[wave][256];   // 512 halves

    // Per-lane constants: combined biases per gate column, layer-0 input weight.
    float b0v[8], b1v[8], w0v[8];
#pragma unroll
    for (int j = 0; j < 8; ++j) {
        int g = 16 * j + lo;
        b0v[j] = bih0[g] + bhh0[g];
        b1v[j] = bih1[g] + bhh1[g];
        w0v[j] = Wih0[g];
    }
    // Row base pointers for x column loads (clamped for tail tiles).
    const float* xr[8];
#pragma unroll
    for (int r = 0; r < 8; ++r) {
        int rr = row0 + r + 8 * hi;
        if (rr > N_NODES - 1) rr = N_NODES - 1;
        xr[r] = x + (long)rr * F_FEAT;
    }

    // State: h as A-fragments (f16) + D-layout f32 copies; c in D-layout f32.
    v16h ha0, ha1;
#pragma unroll
    for (int i = 0; i < 16; ++i) { ha0[i] = (_Float16)0.0f; ha1[i] = (_Float16)0.0f; }
    float c0[16], c1[16], hd0[16], hd1[16];
#pragma unroll
    for (int i = 0; i < 16; ++i) { c0[i] = c1[i] = hd0[i] = hd1[i] = 0.0f; }

    v8f acc[8];

    for (int t = 0; t < F_FEAT; ++t) {
        float xv[8];
#pragma unroll
        for (int r = 0; r < 8; ++r) xv[r] = xr[r][t];

        // ---- layer 0: z = x_t*Wih0^T + h0@Whh0^T + b0 ----
#pragma unroll
        for (int j = 0; j < 8; ++j) {
            v8f cin;
#pragma unroll
            for (int r = 0; r < 8; ++r) cin[r] = b0v[j] + w0v[j] * xv[r];
            const int gb = (16 * j + lo) * 32 + 16 * hi;
            v16h bf = load_frag(&sWhh0[gb], &sWhh0[gb + 8]);
            acc[j] = __builtin_amdgcn_wmma_f32_16x16x32_f16(
                false, ha0, false, bf, (short)0, cin, false, false);
        }
        // gates (PyTorch order i,f,g,o across gate tiles {0,1},{2,3},{4,5},{6,7})
#pragma unroll
        for (int kk = 0; kk < 2; ++kk) {
#pragma unroll
            for (int r = 0; r < 8; ++r) {
                float ig = sigm_f(acc[kk][r]);
                float fg = sigm_f(acc[2 + kk][r]);
                float gg = tanh_f(acc[4 + kk][r]);
                float og = sigm_f(acc[6 + kk][r]);
                int idx = kk * 8 + r;
                float cn = fg * c0[idx] + ig * gg;
                c0[idx] = cn;
                float hn = og * tanh_f(cn);
                hd0[idx] = hn;
                h0stg[(r + 8 * hi) * 32 + 16 * kk + lo] = (_Float16)hn;
            }
        }
        ha0 = load_frag(&h0stg[lo * 32 + 16 * hi], &h0stg[lo * 32 + 16 * hi + 8]);

        // ---- layer 1: z = h0@Wih1^T + h1@Whh1^T + b1 ----
#pragma unroll
        for (int j = 0; j < 8; ++j) {
            v8f cin;
#pragma unroll
            for (int r = 0; r < 8; ++r) cin[r] = b1v[j];
            const int gb = (16 * j + lo) * 32 + 16 * hi;
            v16h bfi = load_frag(&sWih1[gb], &sWih1[gb + 8]);
            v8f z1 = __builtin_amdgcn_wmma_f32_16x16x32_f16(
                false, ha0, false, bfi, (short)0, cin, false, false);
            v16h bfh = load_frag(&sWhh1[gb], &sWhh1[gb + 8]);
            acc[j] = __builtin_amdgcn_wmma_f32_16x16x32_f16(
                false, ha1, false, bfh, (short)0, z1, false, false);
        }
#pragma unroll
        for (int kk = 0; kk < 2; ++kk) {
#pragma unroll
            for (int r = 0; r < 8; ++r) {
                float ig = sigm_f(acc[kk][r]);
                float fg = sigm_f(acc[2 + kk][r]);
                float gg = tanh_f(acc[4 + kk][r]);
                float og = sigm_f(acc[6 + kk][r]);
                int idx = kk * 8 + r;
                float cn = fg * c1[idx] + ig * gg;
                c1[idx] = cn;
                float hn = og * tanh_f(cn);
                hd1[idx] = hn;
                h1stg[(r + 8 * hi) * 32 + 16 * kk + lo] = (_Float16)hn;
            }
        }
        ha1 = load_frag(&h1stg[lo * 32 + 16 * hi], &h1stg[lo * 32 + 16 * hi + 8]);
    }

    // hsum = h0_T + h1_T, written f32 to the per-wave scratch (reuses f16 area).
#pragma unroll
    for (int kk = 0; kk < 2; ++kk) {
#pragma unroll
        for (int r = 0; r < 8; ++r) {
            int m = r + 8 * hi, k = 16 * kk + lo;
            sH[wave][m * 32 + k] = hd0[kk * 8 + r] + hd1[kk * 8 + r];
        }
    }
    __syncthreads();

    // ---- MLP + log_softmax: lanes 0..15 each own one row ----
    if (lane < 16) {
        int row = row0 + lane;
        float hv[32];
#pragma unroll
        for (int k = 0; k < 32; ++k) hv[k] = sH[wave][lane * 32 + k];

        const float* A1s = sMlp;
        const float* b1s = sMlp + 1024;
        const float* A2s = sMlp + 1056;
        const float* b2s = sMlp + 1568;
        const float* A3s = sMlp + 1584;
        const float* b3s = sMlp + 1744;

        float m1[32];
#pragma unroll 4
        for (int j = 0; j < 32; ++j) {
            float s = b1s[j];
            for (int k = 0; k < 32; ++k) s += hv[k] * A1s[k * 32 + j];
            m1[j] = fmaxf(s, 0.0f);
        }
        float m2[16];
#pragma unroll 4
        for (int j = 0; j < 16; ++j) {
            float s = b2s[j];
            for (int k = 0; k < 32; ++k) s += m1[k] * A2s[k * 16 + j];
            m2[j] = fmaxf(s, 0.0f);
        }
        float lg[10];
#pragma unroll
        for (int c = 0; c < 10; ++c) {
            float s = b3s[c];
            for (int k = 0; k < 16; ++k) s += m2[k] * A3s[k * 10 + c];
            lg[c] = s;
        }
        float mx = lg[0];
#pragma unroll
        for (int c = 1; c < 10; ++c) mx = fmaxf(mx, lg[c]);
        float se = 0.0f;
#pragma unroll
        for (int c = 0; c < 10; ++c) se += __expf(lg[c] - mx);
        float lse = mx + __logf(se);
        if (row < N_NODES) {
#pragma unroll
            for (int c = 0; c < 10; ++c) logp[row * 10 + c] = lg[c] - lse;
        }
    }
}

// ---------------------------------------------------------------------------
// support = x @ W1  (11473x128 @ 128x32), WMMA tiled: 4 K-chunks x 2 N-tiles.
// ---------------------------------------------------------------------------
__global__ __launch_bounds__(256) void GCN_gemm_kernel(
    const float* __restrict__ x, const float* __restrict__ W1,
    float* __restrict__ support)
{
    __shared__ _Float16 sW1t[32 * 128];      // transposed: [n][k]
    __shared__ _Float16 sX[8][16 * 128];     // per-wave x tiles (f16)

    const int tid = threadIdx.x;
    for (int i = tid; i < 128 * 32; i += 256) {
        int k = i >> 5, n = i & 31;
        sW1t[n * 128 + k] = (_Float16)W1[i];
    }

    const int wave = tid >> 5;
    const int lane = tid & 31;
    const int lo   = lane & 15;
    const int hi   = lane >> 4;
    const int row0 = (blockIdx.x * 8 + wave) * 16;

    for (int i = lane; i < 16 * 128; i += 32) {
        int m = i >> 7, k = i & 127;
        int rr = row0 + m;
        if (rr > N_NODES - 1) rr = N_NODES - 1;
        sX[wave][i] = (_Float16)x[(long)rr * 128 + k];
    }
    __syncthreads();

    v8f acc[2];
#pragma unroll
    for (int j = 0; j < 2; ++j)
#pragma unroll
        for (int r = 0; r < 8; ++r) acc[j][r] = 0.0f;

#pragma unroll
    for (int q = 0; q < 4; ++q) {
        const int ab = lo * 128 + q * 32 + 16 * hi;
        v16h af = load_frag(&sX[wave][ab], &sX[wave][ab + 8]);
#pragma unroll
        for (int j = 0; j < 2; ++j) {
            const int bb = (16 * j + lo) * 128 + q * 32 + 16 * hi;
            v16h bf = load_frag(&sW1t[bb], &sW1t[bb + 8]);
            acc[j] = __builtin_amdgcn_wmma_f32_16x16x32_f16(
                false, af, false, bf, (short)0, acc[j], false, false);
        }
    }
#pragma unroll
    for (int j = 0; j < 2; ++j) {
#pragma unroll
        for (int r = 0; r < 8; ++r) {
            int m = r + 8 * hi, n = 16 * j + lo;
            int row = row0 + m;
            if (row < N_NODES) support[row * 32 + n] = acc[j][r];
        }
    }
}

__global__ void GCN_zero_kernel(float* __restrict__ p, int n) {
    int i = blockIdx.x * blockDim.x + threadIdx.x;
    if (i < n) p[i] = 0.0f;
}

// One wave32 per edge: scalar edge metadata, coalesced 128B support row,
// global_atomic_add_f32 scatter into gc.
__global__ void GCN_scatter_kernel(
    const int* __restrict__ erow, const int* __restrict__ ecol,
    const float* __restrict__ eval, const float* __restrict__ support,
    float* __restrict__ gc, int E)
{
    int tid = blockIdx.x * blockDim.x + threadIdx.x;
    int e = tid >> 5, f = tid & 31;
    if (e < E) {
        float v = eval[e] * support[(long)ecol[e] * 32 + f];
        atomicAdd(&gc[(long)erow[e] * 32 + f], v);
    }
}

__global__ void GCN_relu_out_kernel(const float* __restrict__ gc,
                                    float* __restrict__ out, int n) {
    int i = blockIdx.x * blockDim.x + threadIdx.x;
    if (i < n) out[i] = fmaxf(gc[i], 0.0f);
}

extern "C" void kernel_launch(void* const* d_in, const int* in_sizes, int n_in,
                              void* d_out, int out_size, void* d_ws, size_t ws_size,
                              hipStream_t stream) {
    const float* x     = (const float*)d_in[0];
    const float* eval  = (const float*)d_in[1];
    const float* W1    = (const float*)d_in[2];
    const float* Wih0  = (const float*)d_in[3];
    const float* Whh0  = (const float*)d_in[4];
    const float* bih0  = (const float*)d_in[5];
    const float* bhh0  = (const float*)d_in[6];
    const float* Wih1  = (const float*)d_in[7];
    const float* Whh1  = (const float*)d_in[8];
    const float* bih1  = (const float*)d_in[9];
    const float* bhh1  = (const float*)d_in[10];
    const float* A1    = (const float*)d_in[11];
    const float* b1    = (const float*)d_in[12];
    const float* A2    = (const float*)d_in[13];
    const float* b2    = (const float*)d_in[14];
    const float* A3    = (const float*)d_in[15];
    const float* b3    = (const float*)d_in[16];
    const int*   erow  = (const int*)d_in[17];
    const int*   ecol  = (const int*)d_in[18];
    const int E = in_sizes[1];

    float* outp    = (float*)d_out;            // logp: N*10, then gc: N*32
    float* support = (float*)d_ws;             // N*32 f32 scratch
    float* gc      = support + N_NODES * HID;  // N*32 f32 scratch

    const int tiles  = (N_NODES + 15) / 16;    // 718
    const int blocks = (tiles + 7) / 8;        // 90 blocks x 8 waves

    // GCN path
    GCN_gemm_kernel<<<blocks, 256, 0, stream>>>(x, W1, support);
    const int gcN = N_NODES * HID;
    GCN_zero_kernel<<<(gcN + 255) / 256, 256, 0, stream>>>(gc, gcN);
    const long sth = (long)E * 32;
    GCN_scatter_kernel<<<(int)((sth + 255) / 256), 256, 0, stream>>>(
        erow, ecol, eval, support, gc, E);
    GCN_relu_out_kernel<<<(gcN + 255) / 256, 256, 0, stream>>>(
        gc, outp + (long)N_NODES * NCLS, gcN);

    // LSTM + MLP path
    GCN_lstm_mlp_kernel<<<blocks, 256, 0, stream>>>(
        x, Wih0, Whh0, bih0, bhh0, Wih1, Whh1, bih1, bhh1,
        A1, b1, A2, b2, A3, b3, outp);
}
